// GAT_61881888801414
// MI455X (gfx1250) — matrix operand order repaired and checked by reading, not compiled, not run on hardware
//
#include <hip/hip_runtime.h>
#include <hip/hip_bf16.h>

typedef __attribute__((ext_vector_type(16))) __bf16 v16bf;
typedef __attribute__((ext_vector_type(8)))  float  v8f;
typedef __attribute__((ext_vector_type(4)))  float  v4f;

#define HC    192
#define NHEAD 3
#define CDIM  64
#define NGRAPH 32

static __device__ __forceinline__ unsigned fkey(float x) {
  unsigned u = __float_as_uint(x);
  return (u & 0x80000000u) ? ~u : (u | 0x80000000u);
}

// ---------------------------------------------------------------------------
// Transpose+convert weights: W[K x 192] fp32  ->  WT[192 x K] bf16
// ---------------------------------------------------------------------------
__global__ void gat_wt_bf16(const float* __restrict__ W, __bf16* __restrict__ WT, int K) {
  int idx = blockIdx.x * 256 + threadIdx.x;   // over 192*K
  if (idx < HC * K) {
    int n = idx / K, k = idx % K;
    WT[idx] = (__bf16)W[k * HC + n];
  }
}

// ---------------------------------------------------------------------------
// C[N x 192] = A[N x K](fp32, cvt->bf16 in-reg) @ B (BT = [192 x K] bf16).
// Block = 128 threads = 4 waves. Wave w owns N-tiles {3w, 3w+1, 3w+2}:
// one A fragment per k-step feeds 3 WMMAs. Row index is clamped (not
// predicated) so the k-loop is divergence-free; stores are row-guarded.
// ---------------------------------------------------------------------------
__global__ void gat_gemm_wmma(const float* __restrict__ A,
                              const __bf16* __restrict__ BT,
                              float* __restrict__ C,
                              int N, int K) {
  const int wave = threadIdx.x >> 5;          // 0..3
  const int lane = threadIdx.x & 31;
  const int tileM = blockIdx.x;
  const int hiHalf = (lane & 16) ? 1 : 0;
  const int rowC = tileM * 16 + (lane & 15);  // true row (store guard)
  int row = rowC;
  if (row > N - 1) row = N - 1;               // clamp: dup rows discarded at store
  const int tileN0 = wave * 3;                // 4 waves * 3 tiles = 12 tiles = 192 cols
  const int colL = lane & 15;

  const float*  aRow = A  + (long)row * K;
  const __bf16* b0p  = BT + (long)((tileN0 + 0) * 16 + colL) * K;
  const __bf16* b1p  = BT + (long)((tileN0 + 1) * 16 + colL) * K;
  const __bf16* b2p  = BT + (long)((tileN0 + 2) * 16 + colL) * K;

  v8f acc0 = {}, acc1 = {}, acc2 = {};

  for (int k0 = 0; k0 < K; k0 += 32) {
    // A fragment (this kernel's contiguous k order, same as B):
    // lanes 0-15 : elems 0..15 -> K=k0+0..15
    // lanes 16-31: elems 0..15 -> K=k0+16..31
    const int kbaseA = k0 + (hiHalf ? 16 : 0);
    v4f a0 = *(const v4f*)(aRow + kbaseA);
    v4f a1 = *(const v4f*)(aRow + kbaseA + 4);
    v4f a2 = *(const v4f*)(aRow + kbaseA + 8);
    v4f a3 = *(const v4f*)(aRow + kbaseA + 12);
    v16bf afrag;
#pragma unroll
    for (int i = 0; i < 4; ++i) {
      afrag[i]      = (__bf16)a0[i];
      afrag[4 + i]  = (__bf16)a1[i];
      afrag[8 + i]  = (__bf16)a2[i];
      afrag[12 + i] = (__bf16)a3[i];
    }
    // B fragments: lanes 0-15 hold K=k0..k0+15 of col, lanes 16-31 K=k0+16..31
    const int kb = k0 + (hiHalf ? 16 : 0);
    v16bf bf0 = *(const v16bf*)(b0p + kb);
    v16bf bf1 = *(const v16bf*)(b1p + kb);
    v16bf bf2 = *(const v16bf*)(b2p + kb);

    acc0 = __builtin_amdgcn_wmma_f32_16x16x32_bf16(false, afrag, false, bf0, (short)0, acc0, false, false);
    acc1 = __builtin_amdgcn_wmma_f32_16x16x32_bf16(false, afrag, false, bf1, (short)0, acc1, false, false);
    acc2 = __builtin_amdgcn_wmma_f32_16x16x32_bf16(false, afrag, false, bf2, (short)0, acc2, false, false);
  }

  // C/D layout: VGPR r -> M = r + (lane>=16 ? 8:0), N = lane&15
  const int crow0 = tileM * 16 + (hiHalf ? 8 : 0);
#pragma unroll
  for (int r = 0; r < 8; ++r) {
    int rr = crow0 + r;
    if (rr < N) {
      float* cp = C + (long)rr * HC + colL;
      cp[(tileN0 + 0) * 16] = acc0[r];
      cp[(tileN0 + 1) * 16] = acc1[r];
      cp[(tileN0 + 2) * 16] = acc2[r];
    }
  }
}

// ---------------------------------------------------------------------------
// al[n,h] = sum_c h[n, h*64+c] * avec[h*64+c]
// ---------------------------------------------------------------------------
__global__ void gat_attn_logits(const float* __restrict__ h,
                                const float* __restrict__ avec,
                                float* __restrict__ al, int N) {
  int idx = blockIdx.x * 256 + threadIdx.x;  // N*3
  if (idx < N * NHEAD) {
    int n = idx / NHEAD, hd = idx % NHEAD;
    const float* hp = h + (long)n * HC + hd * CDIM;
    const float* ap = avec + hd * CDIM;
    float s = 0.f;
#pragma unroll 4
    for (int c = 0; c < CDIM; ++c) s += hp[c] * ap[c];
    al[idx] = s;
  }
}

// ---------------------------------------------------------------------------
// Pass 1 over edges: leaky-relu logit, store, atomicMax segment key.
// ---------------------------------------------------------------------------
__global__ void gat_edge_max(const int* __restrict__ src, const int* __restrict__ dst,
                             const float* __restrict__ als, const float* __restrict__ ald,
                             float* __restrict__ ebuf, unsigned* __restrict__ mkey, int E) {
  int e = blockIdx.x * 256 + threadIdx.x;
  if (e < E) {
    int s = src[e], d = dst[e];
#pragma unroll
    for (int h = 0; h < NHEAD; ++h) {
      float v = als[s * NHEAD + h] + ald[d * NHEAD + h];
      v = v > 0.f ? v : 0.2f * v;
      ebuf[e * NHEAD + h] = v;
      atomicMax(&mkey[d * NHEAD + h], fkey(v));
    }
  }
}

// Decode key -> float max, guard non-finite (empty segments) to 0.
__global__ void gat_node_mfix(const unsigned* __restrict__ mkey, float* __restrict__ m, int n3) {
  int i = blockIdx.x * 256 + threadIdx.x;
  if (i < n3) {
    unsigned k = mkey[i];
    unsigned u = (k & 0x80000000u) ? (k & 0x7FFFFFFFu) : ~k;
    if ((u & 0x7F800000u) == 0x7F800000u) u = 0u;   // inf/nan -> 0.0
    m[i] = __uint_as_float(u);
  }
}

// Pass 2 over edges: ex = exp(e - m[dst]); accumulate denominator.
__global__ void gat_edge_exp(const int* __restrict__ dst, float* __restrict__ ebuf,
                             const float* __restrict__ m, float* __restrict__ den, int E) {
  int e = blockIdx.x * 256 + threadIdx.x;
  if (e < E) {
    int d = dst[e];
#pragma unroll
    for (int h = 0; h < NHEAD; ++h) {
      float ex = __expf(ebuf[e * NHEAD + h] - m[d * NHEAD + h]);
      ebuf[e * NHEAD + h] = ex;
      atomicAdd(&den[d * NHEAD + h], ex);
    }
  }
}

// Pass 3: message scatter. One thread per (edge, 32-channel chunk), 6 chunks.
__global__ void gat_edge_msg(const int* __restrict__ src, const int* __restrict__ dst,
                             const float* __restrict__ ebuf, const float* __restrict__ den,
                             const float* __restrict__ hs, float* __restrict__ out, int E) {
  long t = (long)blockIdx.x * 256 + threadIdx.x;  // E*6
  if (t < (long)E * 6) {
    int e = (int)(t / 6), seg = (int)(t % 6), h = seg >> 1;
    int s = src[e], d = dst[e];
    float alpha = ebuf[e * NHEAD + h] / den[d * NHEAD + h];
    const float* hp = hs + (long)s * HC + seg * 32;
    float* op = out + (long)d * HC + seg * 32;
#pragma unroll
    for (int c = 0; c < 32; ++c) atomicAdd(&op[c], hp[c] * alpha);
  }
}

// bias + relu in place
__global__ void gat_bias_relu(float* __restrict__ x, const float* __restrict__ b, int n192) {
  int i = blockIdx.x * 256 + threadIdx.x;
  if (i < n192) {
    float v = x[i] + b[i % HC];
    x[i] = v > 0.f ? v : 0.f;
  }
}

// mean-pool accumulate
__global__ void gat_pool(const float* __restrict__ x, const int* __restrict__ batch,
                         float* __restrict__ psum, float* __restrict__ pcnt, int N) {
  int t = blockIdx.x * 256 + threadIdx.x;  // N*192
  if (t < N * HC) {
    int n = t / HC, c = t % HC;
    int g = batch[n];
    atomicAdd(&psum[g * HC + c], x[t]);
    if (c == 0) atomicAdd(&pcnt[g], 1.0f);
  }
}

// final linear 576 -> 2, one block of 64 threads (g,o)
__global__ void gat_final(const float* __restrict__ pa, const float* __restrict__ pt,
                          const float* __restrict__ pu, const float* __restrict__ ca,
                          const float* __restrict__ ct, const float* __restrict__ cu,
                          const float* __restrict__ linW, const float* __restrict__ linb,
                          float* __restrict__ out) {
  int t = threadIdx.x;
  if (t < NGRAPH * 2) {
    int g = t >> 1, o = t & 1;
    float s = linb[o];
    for (int j = 0; j < 3 * HC; ++j) {
      int ty = j / HC, c = j % HC;
      const float* ps = ty == 0 ? pa : (ty == 1 ? pt : pu);
      const float* cs = ty == 0 ? ca : (ty == 1 ? ct : cu);
      float cnt = cs[g]; cnt = cnt > 1.f ? cnt : 1.f;
      s += (ps[g * HC + c] / cnt) * linW[j * 2 + o];
    }
    out[g * 2 + o] = s;
  }
}

// ---------------------------------------------------------------------------
static inline int cdiv(long a, long b) { return (int)((a + b - 1) / b); }
static inline size_t al256(size_t x) { return (x + 255) & ~(size_t)255; }

extern "C" void kernel_launch(void* const* d_in, const int* in_sizes, int n_in,
                              void* d_out, int out_size, void* d_ws, size_t ws_size,
                              hipStream_t stream) {
  (void)n_in; (void)out_size; (void)ws_size;
  const float* x_article = (const float*)d_in[0];
  const float* x_tweet   = (const float*)d_in[1];
  const float* x_user    = (const float*)d_in[2];
  const float* Wsrc1 = (const float*)d_in[3];
  const float* Wdst1 = (const float*)d_in[4];
  const float* as1   = (const float*)d_in[5];
  const float* ad1   = (const float*)d_in[6];
  const float* b1    = (const float*)d_in[7];
  const float* Wsrc2 = (const float*)d_in[8];
  const float* Wdst2 = (const float*)d_in[9];
  const float* as2   = (const float*)d_in[10];
  const float* ad2   = (const float*)d_in[11];
  const float* b2    = (const float*)d_in[12];
  const float* linW  = (const float*)d_in[13];
  const float* linb  = (const float*)d_in[14];
  const int* srcs[3] = {(const int*)d_in[15], (const int*)d_in[17], (const int*)d_in[19]};
  const int* dsts[3] = {(const int*)d_in[16], (const int*)d_in[18], (const int*)d_in[20]};
  const int* batch_a = (const int*)d_in[21];
  const int* batch_t = (const int*)d_in[22];
  const int* batch_u = (const int*)d_in[23];

  const int NA = 10000, NT = 100000, NU = 50000;
  const int Es[3] = {in_sizes[15], in_sizes[17], in_sizes[19]};

  // -------- workspace carve-up --------
  char* ws = (char*)d_ws;
  size_t off = 0;
  __bf16* WTs = (__bf16*)(ws + off); off += al256((size_t)HC * 256 * 2);
  __bf16* WTd = (__bf16*)(ws + off); off += al256((size_t)HC * 256 * 2);
  float* hs   = (float*)(ws + off);  off += al256((size_t)NT * HC * 4);
  float* hd   = (float*)(ws + off);  off += al256((size_t)NT * HC * 4);
  float* als  = (float*)(ws + off);  off += al256((size_t)NT * NHEAD * 4);
  float* ald  = (float*)(ws + off);  off += al256((size_t)NT * NHEAD * 4);
  float* ebuf = (float*)(ws + off);  off += al256((size_t)300000 * NHEAD * 4);
  unsigned* mkey = (unsigned*)(ws + off); off += al256((size_t)NT * NHEAD * 4);
  float* mbuf = (float*)(ws + off);  off += al256((size_t)NT * NHEAD * 4);
  float* den  = (float*)(ws + off);  off += al256((size_t)NT * NHEAD * 4);
  float* xa1  = (float*)(ws + off);  off += al256((size_t)NA * HC * 4);
  float* xt1  = (float*)(ws + off);  off += al256((size_t)NT * HC * 4);
  float* xu1  = (float*)(ws + off);  off += al256((size_t)NU * HC * 4);
  float* xa2  = (float*)(ws + off);  off += al256((size_t)NA * HC * 4);
  float* xt2  = (float*)(ws + off);  off += al256((size_t)NT * HC * 4);
  float* xu2  = (float*)(ws + off);  off += al256((size_t)NU * HC * 4);
  float* psum = (float*)(ws + off);  off += al256((size_t)3 * NGRAPH * HC * 4);
  float* pcnt = (float*)(ws + off);  off += al256((size_t)3 * NGRAPH * 4);

  dim3 blk128(128), blk256(256);

  for (int layer = 0; layer < 2; ++layer) {
    const int K = layer == 0 ? 256 : HC;
    const float* Ws = layer == 0 ? Wsrc1 : Wsrc2;
    const float* Wd = layer == 0 ? Wdst1 : Wdst2;
    const float* av = layer == 0 ? as1 : as2;
    const float* bv = layer == 0 ? ad1 : ad2;
    const float* bb = layer == 0 ? b1 : b2;
    const float* XS[3]; const float* XD[3]; float* OUT[3];
    int NS[3], ND[3];
    if (layer == 0) {
      XS[0] = x_tweet; XD[0] = x_article; XS[1] = x_user; XD[1] = x_tweet;
      XS[2] = x_tweet; XD[2] = x_user;
      OUT[0] = xa1; OUT[1] = xt1; OUT[2] = xu1;
    } else {
      XS[0] = xt1; XD[0] = xa1; XS[1] = xu1; XD[1] = xt1; XS[2] = xt1; XD[2] = xu1;
      OUT[0] = xa2; OUT[1] = xt2; OUT[2] = xu2;
    }
    NS[0] = NT; ND[0] = NA; NS[1] = NU; ND[1] = NT; NS[2] = NT; ND[2] = NU;

    for (int t = 0; t < 3; ++t) {
      const int Ns = NS[t], Nd = ND[t], E = Es[t];
      // weight transpose/convert
      gat_wt_bf16<<<cdiv((long)HC * K, 256), blk256, 0, stream>>>(Ws + (long)t * K * HC, WTs, K);
      gat_wt_bf16<<<cdiv((long)HC * K, 256), blk256, 0, stream>>>(Wd + (long)t * K * HC, WTd, K);
      // feature transforms (WMMA bf16): 1 block per 16-row tile, 4 waves x 3 N-tiles
      gat_gemm_wmma<<<dim3(cdiv(Ns, 16)), blk128, 0, stream>>>(XS[t], WTs, hs, Ns, K);
      gat_gemm_wmma<<<dim3(cdiv(Nd, 16)), blk128, 0, stream>>>(XD[t], WTd, hd, Nd, K);
      // attention logits
      gat_attn_logits<<<cdiv((long)Ns * NHEAD, 256), blk256, 0, stream>>>(hs, av + t * HC, als, Ns);
      gat_attn_logits<<<cdiv((long)Nd * NHEAD, 256), blk256, 0, stream>>>(hd, bv + t * HC, ald, Nd);
      // segment softmax + scatter
      hipMemsetAsync(mkey, 0, (size_t)Nd * NHEAD * 4, stream);
      hipMemsetAsync(den, 0, (size_t)Nd * NHEAD * 4, stream);
      hipMemsetAsync(OUT[t], 0, (size_t)Nd * HC * 4, stream);
      gat_edge_max<<<cdiv(E, 256), blk256, 0, stream>>>(srcs[t], dsts[t], als, ald, ebuf, mkey, E);
      gat_node_mfix<<<cdiv((long)Nd * NHEAD, 256), blk256, 0, stream>>>(mkey, mbuf, Nd * NHEAD);
      gat_edge_exp<<<cdiv(E, 256), blk256, 0, stream>>>(dsts[t], ebuf, mbuf, den, E);
      gat_edge_msg<<<cdiv((long)E * 6, 256), blk256, 0, stream>>>(srcs[t], dsts[t], ebuf, den, hs, OUT[t], E);
      gat_bias_relu<<<cdiv((long)Nd * HC, 256), blk256, 0, stream>>>(OUT[t], bb + t * HC, Nd * HC);
    }
  }

  // mean pool + final linear
  hipMemsetAsync(psum, 0, (size_t)3 * NGRAPH * HC * 4, stream);
  hipMemsetAsync(pcnt, 0, (size_t)3 * NGRAPH * 4, stream);
  gat_pool<<<cdiv((long)NA * HC, 256), blk256, 0, stream>>>(xa2, batch_a, psum + 0 * NGRAPH * HC, pcnt + 0 * NGRAPH, NA);
  gat_pool<<<cdiv((long)NT * HC, 256), blk256, 0, stream>>>(xt2, batch_t, psum + 1 * NGRAPH * HC, pcnt + 1 * NGRAPH, NT);
  gat_pool<<<cdiv((long)NU * HC, 256), blk256, 0, stream>>>(xu2, batch_u, psum + 2 * NGRAPH * HC, pcnt + 2 * NGRAPH, NU);
  gat_final<<<1, 64, 0, stream>>>(psum + 0 * NGRAPH * HC, psum + 1 * NGRAPH * HC, psum + 2 * NGRAPH * HC,
                                  pcnt + 0 * NGRAPH, pcnt + 1 * NGRAPH, pcnt + 2 * NGRAPH,
                                  linW, linb, (float*)d_out);
}
